// MetaAdapter_54820962566658
// MI455X (gfx1250) — matrix operand order repaired and checked by reading, not compiled, not run on hardware
//
#include <hip/hip_runtime.h>

// ---------------------------------------------------------------------------
// Fused MetaAdapter for MI455X (gfx1250, wave32, WMMA bf16 16x16x32).
//
// Phase 0 (one-pass converts, memory-bound, ~26MB total):
//   xb   = bf16(x)            (B,T,DIM)        visb = bf16(vision) (B,V,CV)
//   WuT  = bf16(Wu^T) (DIM,CV)   WkT = bf16(Wk^T) (DIM,DIM)   WvT = bf16(Wv^T)
// Phase 1 (all-bf16 WMMA GEMMs, operands staged via async global->LDS):
//   U   = bf16( visb @ Wu + bu )               (V x DIM)
//   Km  = bf16( U @ Wk )                       (V x DIM)
//   Vt  = bf16( U @ Wv ) stored transposed     (DIM x V)
//   Sc  = (xb @ Km^T) / sqrt(DIM)              (T x V)
//   P   = sel ? exp(Sc - max_b) : 0 (bf16), invZ_b        (softmax kernel)
//   out = x + (P @ V') * invZ_b                (T x DIM)
//
// Every GEMM operand is bf16 with K contiguous -> staging is pure
// GLOBAL_LOAD_ASYNC_TO_LDS_B128 (ASYNCcnt), double-buffered (wait <= 8).
// Compute phase per K=32 substep: 12 ds_load_b128 -> 8 back-to-back WMMAs.
// ---------------------------------------------------------------------------

typedef unsigned short u16;
typedef __bf16  bf16x16 __attribute__((ext_vector_type(16)));
typedef u16     u16x16  __attribute__((ext_vector_type(16)));
typedef u16     u16x8   __attribute__((ext_vector_type(8)));
typedef u16     u16x4   __attribute__((ext_vector_type(4)));
typedef float   f32x8   __attribute__((ext_vector_type(8)));
typedef float   f32x4   __attribute__((ext_vector_type(4)));

constexpr int BB = 4, TT = 256, VV = 1024, DIMM = 1024, CVV = 256;
constexpr int BM = 128, BN = 128, BK = 64, NTHR = 256;
constexpr int LDSS = 72;   // LDS row pitch in u16 (144B: 16B-aligned, conflict-free)

enum : int { EP_BF16 = 0, EP_BF16T = 1, EP_F32_SCALE = 2, EP_F32_ADDX = 3 };

__device__ __forceinline__ u16 f2bf(float f) {         // native f32 -> bf16 cvt
    return __builtin_bit_cast(u16, (__bf16)f);
}

// ---- CDNA5 async global->LDS copy (ASYNCcnt), GVS mode: saddr + 32b voffset
__device__ __forceinline__ void async_copy_b128(const void* gbase, int voff_bytes,
                                                void* lds_ptr) {
    unsigned laddr = (unsigned)(unsigned long long)lds_ptr;   // low 32b = LDS offset
    asm volatile("global_load_async_to_lds_b128 %0, %1, %2"
                 :: "v"(laddr), "v"(voff_bytes), "s"(gbase)
                 : "memory");
}
template<int N>
__device__ __forceinline__ void wait_async() {
    asm volatile("s_wait_asynccnt %0" :: "i"(N) : "memory");
}

// A fragment (16x32 bf16): lane<16 -> M=lane, kb=0; lane>=16 -> M=lane-16, kb=8;
// halves[0..7]=K kb..kb+7, halves[8..15]=K kb+16..kb+23.  LDS [m][k], pitch LDSS.
__device__ __forceinline__ bf16x16 frag_a(const u16* l, int row0, int ks, int lane) {
    int r  = row0 + (lane & 15);
    int kb = ks + ((lane & 16) ? 8 : 0);
    const u16* p = l + r * LDSS + kb;
    u16x8 lo = *(const u16x8*)p;
    u16x8 hi = *(const u16x8*)(p + 16);
    u16x16 w = __builtin_shufflevector(lo, hi, 0,1,2,3,4,5,6,7,8,9,10,11,12,13,14,15);
    return __builtin_bit_cast(bf16x16, w);
}

// B fragment (32x16 bf16): lane<16 -> N=lane, kb=0; lane>=16 -> N=lane-16, kb=8;
// halves[0..7]=K kb..kb+7, halves[8..15]=K kb+16..kb+23.  LDS [n][k], pitch LDSS.
__device__ __forceinline__ bf16x16 frag_b(const u16* l, int col0, int ks, int lane) {
    int c  = col0 + (lane & 15);
    int kb = ks + ((lane & 16) ? 8 : 0);
    const u16* p = l + c * LDSS + kb;
    u16x8 lo = *(const u16x8*)p;
    u16x8 hi = *(const u16x8*)(p + 16);
    u16x16 w = __builtin_shufflevector(lo, hi, 0,1,2,3,4,5,6,7,8,9,10,11,12,13,14,15);
    return __builtin_bit_cast(bf16x16, w);
}

// ---------------------------------------------------------------------------
// 128x128-tile all-bf16 GEMM: C = A[M,K] x B[N,K]^T.  256 threads = 8 waves
// (4 along M x 2 along N); wave computes 32x64 = 2x4 f32 accumulators.
// Both operands double-buffered via async global->LDS (8 asyncs/wave/tile).
// ---------------------------------------------------------------------------
template<int EP>
__global__ __launch_bounds__(NTHR)
void gemm128x128(const u16* __restrict__ Ap, long long batchA, int lda,
                 const u16* __restrict__ Bp, long long batchB, int ldb,
                 void*      __restrict__ Cp, long long batchC, int ldc,
                 const float* __restrict__ bias,
                 const float* __restrict__ Xp, long long batchX,
                 const float* __restrict__ invZ,
                 float alpha, int Ktot)
{
    __shared__ u16 lA[2][BM * LDSS];
    __shared__ u16 lB[2][BN * LDSS];

    const int bz   = blockIdx.z;
    const int m0   = blockIdx.y * BM;
    const int n0   = blockIdx.x * BN;
    const int tid  = threadIdx.x;
    const int lane = tid & 31;
    const int wave = tid >> 5;
    const int wm   = wave & 3;     // 4 waves along M -> 32 rows each
    const int wn   = wave >> 2;    // 2 waves along N -> 64 cols each

    const u16* A = Ap + (long long)bz * batchA;
    const u16* B = Bp + (long long)bz * batchB;
    f32x8 acc[2][4] = {};

    auto issue_tile = [&](int kt, int buf) {
        const int k0 = kt * BK;
        u16* dA = &lA[buf][0];
        u16* dB = &lB[buf][0];
        #pragma unroll
        for (int it = 0; it < 4; ++it) {          // 1024 16B chunks of A tile
            int g = it * NTHR + tid;
            int row = g >> 3, c8 = g & 7;
            async_copy_b128(A, (int)(((m0 + row) * lda + k0 + c8 * 8) * 2),
                            dA + row * LDSS + c8 * 8);
        }
        #pragma unroll
        for (int it = 0; it < 4; ++it) {          // 1024 16B chunks of B tile
            int g = it * NTHR + tid;
            int row = g >> 3, c8 = g & 7;
            async_copy_b128(B, (int)(((n0 + row) * ldb + k0 + c8 * 8) * 2),
                            dB + row * LDSS + c8 * 8);
        }
    };

    const int nk = Ktot / BK;
    issue_tile(0, 0);
    for (int kt = 0; kt < nk; ++kt) {
        const int cur = kt & 1;
        if (kt + 1 < nk) {
            issue_tile(kt + 1, cur ^ 1);          // 8 new asyncs outstanding
            wait_async<8>();                      // older tile's 8 have landed
        } else {
            wait_async<0>();
        }
        __syncthreads();

        const u16* bufA = &lA[cur][0];
        const u16* bufB = &lB[cur][0];
        #pragma unroll
        for (int ks = 0; ks < 2; ++ks) {          // two K=32 sub-steps
            // load ALL fragments first (12x ds_load_b128), then 8 WMMAs
            bf16x16 a0 = frag_a(bufA, wm * 32,      ks * 32, lane);
            bf16x16 a1 = frag_a(bufA, wm * 32 + 16, ks * 32, lane);
            bf16x16 bfr[4];
            #pragma unroll
            for (int ns = 0; ns < 4; ++ns)
                bfr[ns] = frag_b(bufB, wn * 64 + ns * 16, ks * 32, lane);
            #pragma unroll
            for (int ns = 0; ns < 4; ++ns) {
                acc[0][ns] = __builtin_amdgcn_wmma_f32_16x16x32_bf16(false, a0, false, bfr[ns],
                                (short)0, acc[0][ns], false, false);
                acc[1][ns] = __builtin_amdgcn_wmma_f32_16x16x32_bf16(false, a1, false, bfr[ns],
                                (short)0, acc[1][ns], false, false);
            }
        }
        __syncthreads();
    }

    // ---- epilogue (C frag: VGPR j -> M=j(+8 for hi lanes), N=lane&15) ----
    const int hi = (lane >> 4) & 1;
    const int cm = m0 + wm * 32 + hi * 8;
    const int cn = n0 + wn * 64 + (lane & 15);

    if constexpr (EP == EP_BF16) {
        u16* C = (u16*)Cp + (long long)bz * batchC;
        #pragma unroll
        for (int ms = 0; ms < 2; ++ms)
            #pragma unroll
            for (int ns = 0; ns < 4; ++ns) {
                int gn = cn + ns * 16;
                float bv = bias ? bias[gn] : 0.0f;
                #pragma unroll
                for (int j = 0; j < 8; ++j)
                    C[(long long)(cm + ms * 16 + j) * ldc + gn] = f2bf(acc[ms][ns][j] + bv);
            }
    } else if constexpr (EP == EP_BF16T) {
        // store transposed: C is [N][M], pitch ldc; 8 contiguous bf16 per lane
        u16* C = (u16*)Cp + (long long)bz * batchC;
        #pragma unroll
        for (int ms = 0; ms < 2; ++ms)
            #pragma unroll
            for (int ns = 0; ns < 4; ++ns) {
                int gn = cn + ns * 16;
                u16x8 h;
                #pragma unroll
                for (int j = 0; j < 8; ++j) h[j] = f2bf(acc[ms][ns][j]);
                *(u16x8*)&C[(long long)gn * ldc + (cm + ms * 16)] = h;
            }
    } else if constexpr (EP == EP_F32_SCALE) {
        float* C = (float*)Cp + (long long)bz * batchC;
        #pragma unroll
        for (int ms = 0; ms < 2; ++ms)
            #pragma unroll
            for (int ns = 0; ns < 4; ++ns) {
                int gn = cn + ns * 16;
                #pragma unroll
                for (int j = 0; j < 8; ++j)
                    C[(long long)(cm + ms * 16 + j) * ldc + gn] = acc[ms][ns][j] * alpha;
            }
    } else {  // EP_F32_ADDX: out = X + acc * invZ[b]
        float* C = (float*)Cp + (long long)bz * batchC;
        const float* X = Xp + (long long)bz * batchX;
        const float iz = invZ[bz];
        #pragma unroll
        for (int ms = 0; ms < 2; ++ms)
            #pragma unroll
            for (int ns = 0; ns < 4; ++ns) {
                int gn = cn + ns * 16;
                #pragma unroll
                for (int j = 0; j < 8; ++j) {
                    long long idx = (long long)(cm + ms * 16 + j) * ldc + gn;
                    C[idx] = X[idx] + acc[ms][ns][j] * iz;
                }
            }
    }
}

// ---------------------------------------------------------------------------
// One-pass f32 -> bf16 elementwise convert (4 elems / thread).
// ---------------------------------------------------------------------------
__global__ __launch_bounds__(NTHR)
void cvt_bf16(const float* __restrict__ in, u16* __restrict__ out)
{
    int i = (blockIdx.x * NTHR + threadIdx.x) * 4;
    f32x4 v = *(const f32x4*)(in + i);
    u16x4 h;
    h[0] = f2bf(v[0]); h[1] = f2bf(v[1]); h[2] = f2bf(v[2]); h[3] = f2bf(v[3]);
    *(u16x4*)(out + i) = h;
}

// ---------------------------------------------------------------------------
// Tiled f32 [K,N] -> bf16 [N,K] transpose-convert (32x32 tiles via LDS).
// ---------------------------------------------------------------------------
__global__ __launch_bounds__(NTHR)
void transpose_cvt(const float* __restrict__ in, u16* __restrict__ out, int K, int N)
{
    __shared__ u16 t[32][33];
    const int n0 = blockIdx.x * 32;
    const int k0 = blockIdx.y * 32;
    const int cx = threadIdx.x & 31;
    const int ry = threadIdx.x >> 5;       // 8 rows per pass
    #pragma unroll
    for (int p = 0; p < 4; ++p) {
        int k = k0 + ry + p * 8;
        t[cx][ry + p * 8] = f2bf(in[(long long)k * N + n0 + cx]);   // t[n][k]
    }
    __syncthreads();
    #pragma unroll
    for (int p = 0; p < 4; ++p) {
        int nl = ry + p * 8;
        out[(long long)(n0 + nl) * K + k0 + cx] = t[nl][cx];
    }
}

// ---------------------------------------------------------------------------
// Per-batch masked softmax prep: max & sum-exp over selected (t,v) cells,
// P[b,t,v] = sel ? exp(Sc - M_b) : 0 (bf16), invZ[b] = 1/Z_b.
// mask = attention_mask (B,V,T) bytes; selected(t,v) = mask[b][v][t].
// ---------------------------------------------------------------------------
__global__ __launch_bounds__(1024)
void softmax_prep(const float* __restrict__ Sc, const unsigned char* __restrict__ mask,
                  u16* __restrict__ P, float* __restrict__ invZ)
{
    __shared__ float red[1024];
    const int b   = blockIdx.x;
    const int tid = threadIdx.x;
    const long long baseS = (long long)b * TT * VV;
    const long long baseM = (long long)b * VV * TT;
    const int N = TT * VV;

    float lm = -3.0e38f;
    for (int i = tid; i < N; i += 1024) {
        int t = i >> 10, v = i & (VV - 1);
        if (mask[baseM + (long long)v * TT + t]) lm = fmaxf(lm, Sc[baseS + i]);
    }
    red[tid] = lm; __syncthreads();
    for (int s = 512; s > 0; s >>= 1) {
        if (tid < s) red[tid] = fmaxf(red[tid], red[tid + s]);
        __syncthreads();
    }
    const float M = red[0];
    __syncthreads();

    float ls = 0.0f;
    for (int i = tid; i < N; i += 1024) {
        int t = i >> 10, v = i & (VV - 1);
        if (mask[baseM + (long long)v * TT + t]) ls += expf(Sc[baseS + i] - M);
    }
    red[tid] = ls; __syncthreads();
    for (int s = 512; s > 0; s >>= 1) {
        if (tid < s) red[tid] += red[tid + s];
        __syncthreads();
    }
    const float Z = red[0];

    for (int i = tid; i < N; i += 1024) {
        int t = i >> 10, v = i & (VV - 1);
        float e = mask[baseM + (long long)v * TT + t] ? expf(Sc[baseS + i] - M) : 0.0f;
        P[baseS + i] = f2bf(e);
    }
    if (tid == 0) invZ[b] = 1.0f / Z;
}

// ---------------------------------------------------------------------------
extern "C" void kernel_launch(void* const* d_in, const int* in_sizes, int n_in,
                              void* d_out, int out_size, void* d_ws, size_t ws_size,
                              hipStream_t stream)
{
    (void)in_sizes; (void)n_in; (void)out_size; (void)ws_size;

    const float*         x      = (const float*)d_in[0];          // (B,T,DIM)
    const float*         vision = (const float*)d_in[1];          // (B,V,CV)
    const unsigned char* amask  = (const unsigned char*)d_in[2];  // (B,V,T) bool
    const float*         Wu     = (const float*)d_in[3];          // (CV,DIM)
    const float*         bu     = (const float*)d_in[4];          // (DIM,)
    const float*         Wk     = (const float*)d_in[5];          // (DIM,DIM)
    const float*         Wv     = (const float*)d_in[6];          // (DIM,DIM)
    float*               out    = (float*)d_out;                  // (B,T,DIM)

    // workspace layout (~38.6 MB)
    u16*   U    = (u16*)d_ws;                                  // B*V*DIM   (8MB)
    u16*   Km   = U    + (size_t)BB * VV * DIMM;               // 8MB
    u16*   Vt   = Km   + (size_t)BB * VV * DIMM;               // V'^T (8MB)
    float* Sc   = (float*)(Vt + (size_t)BB * DIMM * VV);       // B*T*V f32 (4MB)
    u16*   P    = (u16*)(Sc + (size_t)BB * TT * VV);           // 2MB
    float* iZ   = (float*)(P + (size_t)BB * TT * VV);          // 64B (padded)
    u16*   xb   = (u16*)(iZ + 16);                             // bf16(x)    (2MB)
    u16*   visb = xb   + (size_t)BB * TT * DIMM;               // bf16(vision)(2MB)
    u16*   WuT  = visb + (size_t)BB * VV * CVV;                // (DIM,CV)  (0.5MB)
    u16*   WkT  = WuT  + (size_t)DIMM * CVV;                   // (DIM,DIM) (2MB)
    u16*   WvT  = WkT  + (size_t)DIMM * DIMM;                  // (DIM,DIM) (2MB)

    const dim3 blk(NTHR);

    // ---- phase 0: one-pass converts / transposes ----
    cvt_bf16<<<dim3((BB*TT*DIMM)/(NTHR*4)), blk, 0, stream>>>(x, xb);
    cvt_bf16<<<dim3((BB*VV*CVV)/(NTHR*4)),  blk, 0, stream>>>(vision, visb);
    transpose_cvt<<<dim3(DIMM/32, CVV/32),  blk, 0, stream>>>(Wu, WuT, CVV, DIMM);
    transpose_cvt<<<dim3(DIMM/32, DIMM/32), blk, 0, stream>>>(Wk, WkT, DIMM, DIMM);
    transpose_cvt<<<dim3(DIMM/32, DIMM/32), blk, 0, stream>>>(Wv, WvT, DIMM, DIMM);

    // ---- phase 1: all-bf16 async-staged WMMA GEMMs ----
    // 1) U = bf16(visb @ Wu + bu)
    gemm128x128<EP_BF16><<<dim3(DIMM/BN, VV/BM, BB), blk, 0, stream>>>(
        visb, (long long)VV * CVV, CVV, WuT, 0, CVV,
        U, (long long)VV * DIMM, DIMM, bu, nullptr, 0, nullptr, 1.0f, CVV);

    // 2) Km = bf16(U @ Wk)
    gemm128x128<EP_BF16><<<dim3(DIMM/BN, VV/BM, BB), blk, 0, stream>>>(
        U, (long long)VV * DIMM, DIMM, WkT, 0, DIMM,
        Km, (long long)VV * DIMM, DIMM, nullptr, nullptr, 0, nullptr, 1.0f, DIMM);

    // 3) V'^T = bf16(U @ Wv), stored transposed as (DIM x V)
    gemm128x128<EP_BF16T><<<dim3(DIMM/BN, VV/BM, BB), blk, 0, stream>>>(
        U, (long long)VV * DIMM, DIMM, WvT, 0, DIMM,
        Vt, (long long)DIMM * VV, VV, nullptr, nullptr, 0, nullptr, 1.0f, DIMM);

    // 4) Sc = (xb @ Km^T) / 32
    gemm128x128<EP_F32_SCALE><<<dim3(VV/BN, TT/BM, BB), blk, 0, stream>>>(
        xb, (long long)TT * DIMM, DIMM, Km, (long long)VV * DIMM, DIMM,
        Sc, (long long)TT * VV, VV, nullptr, nullptr, 0, nullptr, 0.03125f, DIMM);

    // 5) per-batch masked softmax -> P (bf16), invZ
    softmax_prep<<<dim3(BB), dim3(1024), 0, stream>>>(Sc, amask, P, iZ);

    // 6) out = x + (P @ V') * invZ
    gemm128x128<EP_F32_ADDX><<<dim3(DIMM/BN, TT/BM, BB), blk, 0, stream>>>(
        P, (long long)TT * VV, VV, Vt, (long long)DIMM * VV, VV,
        out, (long long)TT * DIMM, DIMM, nullptr, x, (long long)TT * DIMM, iZ, 1.0f, VV);
}
